// MVKEBlock_10660108828886
// MI455X (gfx1250) — matrix operand order repaired and checked by reading, not compiled, not run on hardware
//
#include <hip/hip_runtime.h>
#include <hip/hip_bf16.h>

typedef __attribute__((ext_vector_type(16))) _Float16 v16h;
typedef __attribute__((ext_vector_type(8)))  float    v8f;
typedef __attribute__((ext_vector_type(4)))  int      v4i;

#define Fdim 200
#define FPAD 224
#define Ddim 32
#define Kdim 4
#define H1d  32
#define H2d  16
#define XS   40      // padded f16 row stride for staged x (80B, 16B aligned, low-conflict)
#define BN_EPS 1e-3f

// ---- optional gfx1250 async global->LDS path (guarded; falls back cleanly) ----
#if defined(__has_builtin)
#if __has_builtin(__builtin_amdgcn_global_load_async_to_lds_b128)
#define HAVE_ASYNC_LDS 1
#endif
#if __has_builtin(__builtin_amdgcn_s_wait_asynccnt)
#define HAVE_WAIT_ASYNC_BUILTIN 1
#endif
#endif

#if defined(HAVE_ASYNC_LDS)
typedef __attribute__((address_space(1))) v4i* gas_v4i_ptr;  // global (AS1)
typedef __attribute__((address_space(3))) v4i* las_v4i_ptr;  // LDS (AS3)
#endif

__device__ __forceinline__ void wait_asynccnt0() {
#if defined(HAVE_WAIT_ASYNC_BUILTIN)
  __builtin_amdgcn_s_wait_asynccnt(0);
#else
  asm volatile("s_wait_asynccnt 0" ::: "memory");
#endif
}

__device__ __forceinline__ float wave_rmax(float v) {
#pragma unroll
  for (int o = 16; o; o >>= 1) v = fmaxf(v, __shfl_xor(v, o, 32));
  return v;
}
__device__ __forceinline__ float wave_rsum(float v) {
#pragma unroll
  for (int o = 16; o; o >>= 1) v += __shfl_xor(v, o, 32);
  return v;
}

__global__ __launch_bounds__(256)
void mvke_fused(const float* __restrict__ x, const float* __restrict__ tag,
                const float* __restrict__ vk,
                const float* __restrict__ W1, const float* __restrict__ b1,
                const float* __restrict__ g1, const float* __restrict__ be1,
                const float* __restrict__ m1, const float* __restrict__ v1,
                const float* __restrict__ W2, const float* __restrict__ b2,
                const float* __restrict__ g2, const float* __restrict__ be2,
                const float* __restrict__ m2, const float* __restrict__ v2,
                float* __restrict__ out)
{
  __shared__ _Float16 xs[FPAD * XS];          // staged x[b], f16, zero-padded rows
#if defined(HAVE_ASYNC_LDS)
  __shared__ float    xstage[Fdim * Ddim];    // raw f32 landing zone for async copy
#endif
  __shared__ _Float16 vkh[Ddim * Kdim];
  __shared__ float    vkf[Ddim * Kdim];
  __shared__ float    tagv[Ddim];
  __shared__ float    scoresS[Kdim * FPAD];   // scores then exp()
  __shared__ _Float16 probsh[Kdim * FPAD];    // softmax probs, f16, zero pad f>=200
  __shared__ float    part[8 * 4 * 16];       // per-wave partial context tiles
  __shared__ float    ctx[Kdim * Ddim];
  __shared__ float    w1t[Kdim * H1d * 33];   // W1 transposed [k][h][d], stride 33
  __shared__ float    w2t[Kdim * H2d * 33];   // W2 transposed [k][j][h], stride 33
  __shared__ float    a1s[Kdim * H1d], c1s[Kdim * H1d];
  __shared__ float    a2s[Kdim * H2d], c2s[Kdim * H2d];
  __shared__ float    h1s[Kdim * H1d];
  __shared__ float    h2s[Kdim * H2d];
  __shared__ float    gatep[Kdim];

  const int b    = blockIdx.x;
  const int tid  = threadIdx.x;
  const int wave = tid >> 5;
  const int lane = tid & 31;
  const float scale = 0.1767766952966369f;    // 1/sqrt(32)

  // ---------------- phase 0: stage everything into LDS ----------------
  const float* xb = x + (size_t)b * (Fdim * Ddim);

#if defined(HAVE_ASYNC_LDS)
  // async DMA-style copy: global -> LDS without VGPR round trip (ASYNCcnt path)
  for (int i = tid; i < (Fdim * Ddim) / 4; i += 256) {
    __builtin_amdgcn_global_load_async_to_lds_b128(
        (gas_v4i_ptr)(xb + i * 4),
        (las_v4i_ptr)(&xstage[i * 4]),
        0, 0);
  }
#endif

  if (tid < Ddim * Kdim) { float v = vk[tid]; vkf[tid] = v; vkh[tid] = (_Float16)v; }
  if (tid < Ddim) tagv[tid] = tag[(size_t)b * Ddim + tid];
  for (int i = tid; i < Kdim * Ddim * H1d; i += 256) {
    int k = i >> 10, d = (i >> 5) & 31, h = i & 31;
    w1t[(k * H1d + h) * 33 + d] = W1[i];
  }
  for (int i = tid; i < Kdim * H1d * H2d; i += 256) {
    int k = i >> 9, h = (i >> 4) & 31, j = i & 15;
    w2t[(k * H2d + j) * 33 + h] = W2[i];
  }
  if (tid < Kdim * H1d) {
    float a = g1[tid] * rsqrtf(v1[tid] + BN_EPS);
    a1s[tid] = a;
    c1s[tid] = (b1[tid] - m1[tid]) * a + be1[tid];   // bias folded into BN shift
  } else if (tid < Kdim * H1d + Kdim * H2d) {
    int i = tid - Kdim * H1d;
    float a = g2[i] * rsqrtf(v2[i] + BN_EPS);
    a2s[i] = a;
    c2s[i] = (b2[i] - m2[i]) * a + be2[i];
  }

#if defined(HAVE_ASYNC_LDS)
  wait_asynccnt0();      // this wave's async copies landed in LDS
  __syncthreads();       // all waves' copies landed
  for (int i = tid; i < FPAD * 8; i += 256) {   // f32 LDS -> f16 LDS (padded)
    int f = i >> 3, j = (i & 7) * 4;
    float4 v = make_float4(0.f, 0.f, 0.f, 0.f);
    if (f < Fdim) v = *(const float4*)(&xstage[f * Ddim + j]);
    _Float16* p = &xs[f * XS + j];
    p[0] = (_Float16)v.x; p[1] = (_Float16)v.y;
    p[2] = (_Float16)v.z; p[3] = (_Float16)v.w;
  }
#else
  for (int i = tid; i < FPAD * 8; i += 256) {
    int f = i >> 3, j = (i & 7) * 4;
    float4 v = make_float4(0.f, 0.f, 0.f, 0.f);
    if (f < Fdim) v = *(const float4*)(xb + f * Ddim + j);
    _Float16* p = &xs[f * XS + j];
    p[0] = (_Float16)v.x; p[1] = (_Float16)v.y;
    p[2] = (_Float16)v.z; p[3] = (_Float16)v.w;
  }
#endif
  __syncthreads();

  // ---------------- phase 1: scores = x @ vk  (WMMA f16 -> f32) ----------------
  {
    const int n  = lane & 15;             // N = k column
    const int kb = (lane >> 4) * 16;      // K block for B (lane half)
    v16h bf;
#pragma unroll
    for (int j = 0; j < 16; ++j)
      bf[j] = (n < Kdim) ? vkh[(kb + j) * Kdim + n] : (_Float16)0.f;

    for (int t = wave; t < 13; t += 8) {  // 13 tiles of 16 f-rows (208 >= 200)
      int f0 = t * 16;
      const _Float16* xr = &xs[(f0 + (lane & 15)) * XS];
      int kb0 = (lane >> 4) * 16;         // f16 A: K blocks of 16 per lane half
      v16h af;
#pragma unroll
      for (int j = 0; j < 8; ++j) { af[j] = xr[kb0 + j]; af[8 + j] = xr[8 + kb0 + j]; }
      v8f c = {};
      c = __builtin_amdgcn_wmma_f32_16x16x32_f16(false, af, false, bf,
                                                 (short)0, c, false, false);
      if (n < Kdim) {                     // C: M = r + 8*(lane>>4), N = lane&15
        int fb = f0 + ((lane >> 4) << 3);
#pragma unroll
        for (int r = 0; r < 8; ++r) scoresS[n * FPAD + fb + r] = c[r];
      }
    }
  }
  __syncthreads();

  // ---------------- phase 2: softmax over f per k; gate softmax ----------------
  if (wave < 4) {
    int k = wave;
    float m = -3.0e38f;
    for (int f = lane; f < Fdim; f += 32) m = fmaxf(m, scoresS[k * FPAD + f]);
    m = wave_rmax(m);
    float s = 0.f;
    for (int f = lane; f < Fdim; f += 32) {
      float e = __expf(scoresS[k * FPAD + f] - m);
      scoresS[k * FPAD + f] = e;
      s += e;
    }
    s = wave_rsum(s);
    float inv = 1.f / s;
    for (int f = lane; f < FPAD; f += 32)
      probsh[k * FPAD + f] = (_Float16)((f < Fdim) ? scoresS[k * FPAD + f] * inv : 0.f);
  } else if (wave == 4 && lane < Kdim) {
    float gs = 0.f;
#pragma unroll
    for (int d = 0; d < Ddim; ++d) gs += tagv[d] * vkf[d * Kdim + lane];
    gs *= scale;
    float m = fmaxf(gs, __shfl_xor(gs, 1, 4));
    m = fmaxf(m, __shfl_xor(m, 2, 4));
    float e = __expf(gs - m);
    float s = e + __shfl_xor(e, 1, 4);
    s += __shfl_xor(s, 2, 4);
    gatep[lane] = e / s;
  }
  __syncthreads();

  // ---------------- phase 3: context = (probs @ x) * scale  (WMMA) ----------------
  {
    const int nt  = wave >> 2;            // which 16-wide d tile (0/1)
    const int g   = wave & 3;             // K-chunk group within tile
    const int d0  = nt * 16;
    const int kr  = lane & 15;            // A: M = k row
    const int kb0 = (lane >> 4) * 8;
    const int col = d0 + (lane & 15);     // B: N = d column
    const int kb  = (lane >> 4) * 16;
    v8f cc = {};
    for (int ch = g; ch < 7; ch += 4) {   // 7 K-chunks of 32 f (224, zero padded)
      int f0 = ch * 32;
      v16h pa;
#pragma unroll
      for (int j = 0; j < 16; ++j) pa[j] = (_Float16)0.f;
      if (kr < Kdim) {
        const _Float16* pr = &probsh[kr * FPAD + f0];
#pragma unroll
        for (int j = 0; j < 8; ++j) { pa[j] = pr[kb + j]; pa[8 + j] = pr[8 + kb + j]; }
      }
      v16h xbf;
      const _Float16* xc = &xs[(f0 + kb) * XS + col];
#pragma unroll
      for (int j = 0; j < 16; ++j) xbf[j] = xc[j * XS];
      cc = __builtin_amdgcn_wmma_f32_16x16x32_f16(false, pa, false, xbf,
                                                  (short)0, cc, false, false);
    }
    if (lane < 16) {                      // valid M rows: k = r (< 4), lane half 0
#pragma unroll
      for (int r = 0; r < 4; ++r) part[(wave * 4 + r) * 16 + lane] = cc[r];
    }
  }
  __syncthreads();
  if (tid < Kdim * Ddim) {                // deterministic cross-wave reduce
    int k = tid >> 5, d = tid & 31;
    int nt = d >> 4, n = d & 15;
    float s = 0.f;
#pragma unroll
    for (int g = 0; g < 4; ++g) s += part[((nt * 4 + g) * 4 + k) * 16 + n];
    ctx[k * Ddim + d] = s * scale;
  }
  __syncthreads();

  // ---------------- phase 4: expert towers (VALU; tiny) ----------------
  if (tid < Kdim * H1d) {
    int k = tid >> 5;
    const float* w  = &w1t[tid * 33];     // (k*H1 + h)*33, tid == k*32+h
    const float* cr = &ctx[k * Ddim];
    float acc = 0.f;
#pragma unroll
    for (int d = 0; d < Ddim; ++d) acc = fmaf(cr[d], w[d], acc);
    h1s[tid] = fmaxf(a1s[tid] * acc + c1s[tid], 0.f);
  }
  __syncthreads();
  if (tid < Kdim * H2d) {
    int k = tid >> 4;
    const float* w  = &w2t[tid * 33];     // (k*H2 + j)*33, tid == k*16+j
    const float* hr = &h1s[k * H1d];
    float acc = 0.f;
#pragma unroll
    for (int h = 0; h < H1d; ++h) acc = fmaf(hr[h], w[h], acc);
    h2s[tid] = fmaxf(a2s[tid] * acc + c2s[tid], 0.f);
  }
  __syncthreads();
  if (tid < H2d) {
    float acc = 0.f;
#pragma unroll
    for (int k = 0; k < Kdim; ++k) acc += h2s[k * H2d + tid] * gatep[k];
    out[(size_t)b * H2d + tid] = acc;
  }
}

extern "C" void kernel_launch(void* const* d_in, const int* in_sizes, int n_in,
                              void* d_out, int out_size, void* d_ws, size_t ws_size,
                              hipStream_t stream) {
  (void)in_sizes; (void)n_in; (void)d_ws; (void)ws_size; (void)out_size;
  const float* x   = (const float*)d_in[0];
  const float* tg  = (const float*)d_in[1];
  const float* vk  = (const float*)d_in[2];
  const float* W1  = (const float*)d_in[3];
  const float* b1  = (const float*)d_in[4];
  const float* g1  = (const float*)d_in[5];
  const float* be1 = (const float*)d_in[6];
  const float* m1  = (const float*)d_in[7];
  const float* v1  = (const float*)d_in[8];
  const float* W2  = (const float*)d_in[9];
  const float* b2  = (const float*)d_in[10];
  const float* g2  = (const float*)d_in[11];
  const float* be2 = (const float*)d_in[12];
  const float* m2  = (const float*)d_in[13];
  const float* v2  = (const float*)d_in[14];
  float* out = (float*)d_out;

  mvke_fused<<<16384, 256, 0, stream>>>(x, tg, vk, W1, b1, g1, be1, m1, v1,
                                        W2, b2, g2, be2, m2, v2, out);
}
